// ParameterizedExperts_8383776161863
// MI455X (gfx1250) — matrix operand order repaired and checked by reading, not compiled, not run on hardware
//
#include <hip/hip_runtime.h>

// ---------------------------------------------------------------------------
// Balanced MoE grouped GEMM for MI455X (gfx1250, wave32, WMMA).
//   out[e*TPE + t, :] = input[e*TPE + t, :] @ weight[e].T + bias[e]
// E=64 experts, D_IN = D_OUT = 512, TPE = 4096 tokens per expert.
//
// Memory-bound design point: 1.09 GB of mandatory HBM traffic (~47 us at
// 23.3 TB/s) vs ~137 GFLOP. f16 WMMA (f32 accumulate) comfortably exceeds
// the ~2.9 PFLOP/s needed to stay under the bandwidth floor; fp32 WMMA
// (16x16x4) would not. So: convert fp32 -> f16 in the LDS staging path,
// accumulate in f32 via v_wmma_f32_16x16x32_f16.
// ---------------------------------------------------------------------------

typedef _Float16 h16 __attribute__((ext_vector_type(16)));
typedef _Float16 h8  __attribute__((ext_vector_type(8)));
typedef _Float16 h4  __attribute__((ext_vector_type(4)));
typedef float    f8  __attribute__((ext_vector_type(8)));
typedef float    f4  __attribute__((ext_vector_type(4)));

#define NUM_EXPERTS 64
#define D_IN        512
#define D_OUT       512
#define TPE         4096   // tokens per expert (T / E, balanced routing)

#define BM 128             // workgroup tile M (tokens)
#define BN 128             // workgroup tile N (out features)
#define BK 32              // K step = WMMA K for f16
#define LDS_STRIDE 40      // halves per LDS row (32 + 8 pad -> bank spread)
#define KSTEPS (D_IN / BK) // 16

__global__ __launch_bounds__(256)
void moe_grouped_gemm_wmma(const float* __restrict__ input,
                           const float* __restrict__ weight,
                           const float* __restrict__ bias,
                           float* __restrict__ out)
{
    // 10 KB + 10 KB staging (f16), double-barrier pipelined.
    __shared__ __align__(16) _Float16 sA[BM * LDS_STRIDE];
    __shared__ __align__(16) _Float16 sB[BN * LDS_STRIDE];

    const int tid    = threadIdx.x;
    const int lane   = tid & 31;
    const int wave   = tid >> 5;
    const int laneN  = lane & 15;   // row/col selector within 16
    const int laneHi = lane >> 4;   // 0 = lanes 0-15, 1 = lanes 16-31

    const int nTile = blockIdx.x;   // 0..3   (N tiles of 128)
    const int mTile = blockIdx.y;   // 0..31  (M tiles of 128)
    const int e     = blockIdx.z;   // 0..63  (expert)

    const int waveM = (wave & 3) * 32;   // 4 waves along M -> 32 rows each
    const int waveN = (wave >> 2) * 64;  // 2 waves along N -> 64 cols each

    // ---- global -> register staging mapping: each thread owns 16 floats of
    //      each tile (half of one 32-float row), 256 threads cover 128x32.
    const int sRow = tid >> 1;           // 0..127
    const int sCol = (tid & 1) * 16;     // 0 or 16

    const size_t aBase =
        ((size_t)(e * TPE + mTile * BM + sRow)) * D_IN + sCol;
    const size_t bBase =
        ((size_t)e * D_OUT + (size_t)(nTile * BN + sRow)) * D_IN + sCol;

    f4 ra[4], rb[4];

    auto gload = [&](int kt) {
        const float* pa = input  + aBase + (size_t)kt * BK;
        const float* pb = weight + bBase + (size_t)kt * BK;
#pragma unroll
        for (int j = 0; j < 4; ++j) {
            ra[j] = *reinterpret_cast<const f4*>(pa + 4 * j);
            rb[j] = *reinterpret_cast<const f4*>(pb + 4 * j);
        }
    };

    auto lstore = [&]() {
#pragma unroll
        for (int j = 0; j < 4; ++j) {
            h4 ha, hb;
#pragma unroll
            for (int c = 0; c < 4; ++c) {
                ha[c] = (_Float16)ra[j][c];
                hb[c] = (_Float16)rb[j][c];
            }
            *reinterpret_cast<h4*>(&sA[sRow * LDS_STRIDE + sCol + 4 * j]) = ha;
            *reinterpret_cast<h4*>(&sB[sRow * LDS_STRIDE + sCol + 4 * j]) = hb;
        }
    };

    // ---- A fragment (16x32 f16). CDNA5 layout: lanes 0-15 hold K 0-7 &
    //      16-23; lanes 16-31 hold K 8-15 & 24-31. Two ds_load_b128 each.
    auto loadA = [&](int mi) -> h16 {
        const int r  = waveM + mi * 16 + laneN;
        const int k0 = laneHi * 8;
        h8 lo = *reinterpret_cast<const h8*>(&sA[r * LDS_STRIDE + k0]);
        h8 hi = *reinterpret_cast<const h8*>(&sA[r * LDS_STRIDE + k0 + 16]);
        return __builtin_shufflevector(lo, hi, 0, 1, 2, 3, 4, 5, 6, 7,
                                               8, 9, 10, 11, 12, 13, 14, 15);
    };

    // ---- B fragment (32x16 f16): lane = column N; lanes 0-15 hold K 0-7 &
    //      16-23; lanes 16-31 hold K 8-15 & 24-31. weight rows are [N][K]
    //      so LDS tile needs no transpose.
    auto loadB = [&](int ni) -> h16 {
        const int r  = waveN + ni * 16 + laneN;
        const int k0 = laneHi * 8;
        h8 lo = *reinterpret_cast<const h8*>(&sB[r * LDS_STRIDE + k0]);
        h8 hi = *reinterpret_cast<const h8*>(&sB[r * LDS_STRIDE + k0 + 16]);
        return __builtin_shufflevector(lo, hi, 0, 1, 2, 3, 4, 5, 6, 7,
                                               8, 9, 10, 11, 12, 13, 14, 15);
    };

    f8 acc[2][4];
#pragma unroll
    for (int mi = 0; mi < 2; ++mi)
#pragma unroll
        for (int ni = 0; ni < 4; ++ni)
            acc[mi][ni] = (f8){0.f, 0.f, 0.f, 0.f, 0.f, 0.f, 0.f, 0.f};

    gload(0);

#pragma unroll 1
    for (int kt = 0; kt < KSTEPS; ++kt) {
        __syncthreads();           // previous tile fully consumed
        lstore();                  // fp32 -> f16 convert into LDS
        __syncthreads();           // tile visible to all waves

        if (kt + 1 < KSTEPS)       // prefetch next tile; latency hides
            gload(kt + 1);         // behind the DS loads + WMMAs below

        h16 afrag[2], bfrag[4];
#pragma unroll
        for (int mi = 0; mi < 2; ++mi) afrag[mi] = loadA(mi);
#pragma unroll
        for (int ni = 0; ni < 4; ++ni) bfrag[ni] = loadB(ni);

#pragma unroll
        for (int mi = 0; mi < 2; ++mi)
#pragma unroll
            for (int ni = 0; ni < 4; ++ni)
                acc[mi][ni] = __builtin_amdgcn_wmma_f32_16x16x32_f16(
                    /*neg_a=*/false, afrag[mi],
                    /*neg_b=*/false, bfrag[ni],
                    /*c_mod=*/(short)0, acc[mi][ni],
                    /*reuse_a=*/false, /*reuse_b=*/false);
    }

    // ---- epilogue: C layout = lane%16 -> N, vgpr r -> M = r + 8*(lane>=16)
    const int colBase = nTile * BN + waveN;
#pragma unroll
    for (int ni = 0; ni < 4; ++ni) {
        const int col = colBase + ni * 16 + laneN;
        const float bv = bias[e * D_OUT + col];
#pragma unroll
        for (int mi = 0; mi < 2; ++mi) {
            const int rowBase = e * TPE + mTile * BM + waveM + mi * 16 + laneHi * 8;
#pragma unroll
            for (int r = 0; r < 8; ++r) {
                out[(size_t)(rowBase + r) * D_OUT + col] = acc[mi][ni][r] + bv;
            }
        }
    }
}

extern "C" void kernel_launch(void* const* d_in, const int* in_sizes, int n_in,
                              void* d_out, int out_size, void* d_ws, size_t ws_size,
                              hipStream_t stream) {
    (void)in_sizes; (void)n_in; (void)out_size; (void)d_ws; (void)ws_size;
    const float* input  = (const float*)d_in[0];
    // d_in[1] = num_experts_per_token (int64): balanced routing, static here.
    const float* weight = (const float*)d_in[2];
    const float* bias   = (const float*)d_in[3];
    float* out = (float*)d_out;

    dim3 grid(D_OUT / BN, TPE / BM, NUM_EXPERTS);   // (4, 32, 64)
    dim3 block(256);
    moe_grouped_gemm_wmma<<<grid, block, 0, stream>>>(input, weight, bias, out);
}
